// Memory_Transformer_49134425866265
// MI455X (gfx1250) — compile-verified, matrix operand order checked by reading, not run on hardware
//
#include <hip/hip_runtime.h>
#include <hip/hip_bf16.h>
#include <stdint.h>

#define DK 256
#define DV 256
#define NSEG 8

typedef _Float16 h16;
typedef __attribute__((ext_vector_type(16))) _Float16 v16h;
typedef __attribute__((ext_vector_type(8)))  _Float16 v8h;
typedef __attribute__((ext_vector_type(4)))  _Float16 v4h;
typedef __attribute__((ext_vector_type(8)))  float    v8f;

typedef __attribute__((ext_vector_type(4))) unsigned int u32x4;
typedef __attribute__((ext_vector_type(8))) int          i32x8;
typedef __attribute__((ext_vector_type(4))) int          i32x4;

#if __has_builtin(__builtin_amdgcn_tensor_load_to_lds)
#define USE_TDM 1
#endif

// Assemble a 16x32 f16 A/B fragment from two contiguous 16B chunks:
//   halves[0..7]  = K h*8+0..7, halves[8..15] = K 16+h*8+0..7 (per ISA layout)
__device__ __forceinline__ v16h make_frag(v8h lo, v8h hi) {
    v16h r;
#pragma unroll
    for (int i = 0; i < 8; ++i) { r[i] = lo[i]; r[i + 8] = hi[i]; }
    return r;
}

// generic (flat) LDS pointer -> workgroup-relative LDS byte offset
__device__ __forceinline__ uint32_t lds_offset(const void* p) {
    return (uint32_t)(uintptr_t)p;   // flat LDS aperture: addr[31:0] is the LDS offset
}

#ifdef USE_TDM
// ---------------------------------------------------------------------------
// Issue one TDM 2-D tile load (CDNA5 ISA).
//   X dim: tile0 elements (2B each) along tensor_dim0 (OOB reads return 0)
//   Y dim: tile1 rows, row stride = stride0 elements
// This toolchain's builtin is the 6-operand form:
//   (u32x4 group0, i32x8 group1, i32x4 group2, i32x4 group3, i32x8 extra, i32 cpol)
// groups 2/3 and the trailing i32x8 are zero (<=2-D tensor, probe-verified form).
// ---------------------------------------------------------------------------
__device__ __forceinline__ void tdm_load_2d(uint32_t lds_off, const void* gptr,
                                            uint32_t tdim0, uint32_t tdim1,
                                            uint32_t tile0, uint32_t tile1,
                                            uint64_t stride0) {
    const uint64_t ga = (uint64_t)(uintptr_t)gptr;

    u32x4 g0;
    g0[0] = 1u;                                                  // count=1, user D#
    g0[1] = lds_off;                                             // lds_addr (bytes)
    g0[2] = (uint32_t)(ga & 0xFFFFFFFFu);                        // global_addr[31:0]
    g0[3] = (uint32_t)((ga >> 32) & 0x01FFFFFFu) | (2u << 30);   // global_addr[56:32] | type=2

    // group1 (256b): mask=0, data_size=1 (2B), no barrier/iterate/pad
    const uint64_t qw0 = (1ull << 16) | ((uint64_t)(tdim0 & 0xFFFFu) << 48);
    const uint64_t qw1 = (uint64_t)((tdim0 >> 16) & 0xFFFFu)
                       | ((uint64_t)tdim1 << 16)
                       | ((uint64_t)(tile0 & 0xFFFFu) << 48);
    const uint64_t qw2 = (uint64_t)(tile1 & 0xFFFFu)                       // tile_dim1, tile_dim2=0
                       | ((stride0 & 0xFFFFFFFFull) << 32);                // dim0_stride[31:0]
    const uint64_t qw3 = (stride0 >> 32) & 0xFFFFull;                      // dim0_stride[47:32], dim1_stride=0
    i32x8 g1;
    g1[0] = (int)(uint32_t)qw0; g1[1] = (int)(uint32_t)(qw0 >> 32);
    g1[2] = (int)(uint32_t)qw1; g1[3] = (int)(uint32_t)(qw1 >> 32);
    g1[4] = (int)(uint32_t)qw2; g1[5] = (int)(uint32_t)(qw2 >> 32);
    g1[6] = (int)(uint32_t)qw3; g1[7] = (int)(uint32_t)(qw3 >> 32);

    i32x4 gz4 = {0, 0, 0, 0};                 // groups 2/3 unused (<=2D tensor)
    i32x8 gz8 = {0, 0, 0, 0, 0, 0, 0, 0};     // trailing operand (zero-filled)
    __builtin_amdgcn_tensor_load_to_lds(g0, g1, gz4, gz4, gz8, 0);
}
#endif

// ---------------------------------------------------------------------------
// Pre-pass: q (f32) -> Qh (f16)
// ---------------------------------------------------------------------------
__global__ void build_q_f16(const float* __restrict__ q, h16* __restrict__ Qh, int nElem) {
    int flat = (blockIdx.x * blockDim.x + threadIdx.x) * 4;
    if (flat >= nElem) return;
    float4 x = *(const float4*)(q + flat);
    v4h y;
    y[0] = (h16)x.x; y[1] = (h16)x.y; y[2] = (h16)x.z; y[3] = (h16)x.w;
    *(v4h*)(Qh + flat) = y;
}

// ---------------------------------------------------------------------------
// Pre-pass: merged K = concat(K_cache[:old], k) -> Kh (f16, row-major)
// ---------------------------------------------------------------------------
__global__ void build_k_f16(const float* __restrict__ Kc, const float* __restrict__ knew,
                            const int* __restrict__ old_ptr, h16* __restrict__ Kh, int Bq) {
    const int old  = *old_ptr;
    const int newN = old + Bq;
    int flat = (blockIdx.x * blockDim.x + threadIdx.x) * 4;
    if (flat >= newN * DK) return;
    int row = flat / DK;
    int col = flat - row * DK;
    const float* src = (row < old) ? (Kc + (size_t)row * DK + col)
                                   : (knew + (size_t)(row - old) * DK + col);
    float4 x = *(const float4*)src;
    v4h y;
    y[0] = (h16)x.x; y[1] = (h16)x.y; y[2] = (h16)x.z; y[3] = (h16)x.w;
    *(v4h*)(Kh + flat) = y;
}

// ---------------------------------------------------------------------------
// Pre-pass: merged V transposed -> VtG[col][key] (f16), LDS-tiled transpose
// ---------------------------------------------------------------------------
__global__ void build_vt_f16(const float* __restrict__ Vc, const float* __restrict__ vnew,
                             const int* __restrict__ old_ptr, h16* __restrict__ Vt, int Bq) {
    __shared__ float tile[16][17];
    const int old  = *old_ptr;
    const int newN = old + Bq;
    const int kb = blockIdx.x * 16;
    const int cb = blockIdx.y * 16;
    const int tx = threadIdx.x, ty = threadIdx.y;

    int row = kb + ty, col = cb + tx;
    float val = 0.f;
    if (row < newN)
        val = (row < old) ? Vc[(size_t)row * DV + col]
                          : vnew[(size_t)(row - old) * DV + col];
    tile[ty][tx] = val;
    __syncthreads();

    int okey = kb + tx;
    int ocol = cb + ty;
    if (okey < newN)
        Vt[(size_t)ocol * newN + okey] = (h16)tile[tx][ty];
}

// ---------------------------------------------------------------------------
// Flash-decoding main kernel.
// grid = (Bq/128, NSEG), block = 256 threads (8 waves).
// K/V 32-key tiles staged into double-buffered LDS by the Tensor Data Mover
// (one 1-D K descriptor + one 2-D strided V descriptor per tile, issued by
// wave 0, TENSORcnt-tracked) and shared by all 8 waves.
// ---------------------------------------------------------------------------
__global__ __launch_bounds__(256) void attn_flash_wmma(
    const h16* __restrict__ Qh, const h16* __restrict__ Kh, const h16* __restrict__ VtG,
    const int* __restrict__ old_ptr, float* __restrict__ Opart, float* __restrict__ ML,
    int Bq) {

    __shared__ h16 Kt [2][32 * DK];   // 32 keys x 256 feats, row-major (16KB each)
    __shared__ h16 Vts[2][DV * 32];   // 256 cols x 32 keys (16KB each)
    __shared__ h16 Pst[8][16][32];    // per-wave P staging (1KB each)

    const int tid  = threadIdx.x;
    const int lane = tid & 31;
    const int wave = tid >> 5;
    const int n    = lane & 15;
    const int h    = lane >> 4;
    const int h8   = h * 8;

    const int old  = *old_ptr;
    const int newN = old + Bq;
    const int seg  = blockIdx.y;
    const int q0   = (blockIdx.x * 8 + wave) * 16;

    // segment bounds (32-aligned), clamped by the block-uniform causal bound
    const int segLen = (((newN + NSEG - 1) / NSEG) + 31) & ~31;
    const int k_lo   = seg * segLen;
    const int kend_b = old + (blockIdx.x + 1) * 128;
    int k_hi = (seg + 1) * segLen;
    if (k_hi > kend_b) k_hi = kend_b;
    const int nIter = (k_hi > k_lo) ? ((k_hi - k_lo + 31) >> 5) : 0;

    // resident Q fragments (8 chunks of the 256-wide K-dim)
    v16h qf[8];
#pragma unroll
    for (int c = 0; c < 8; ++c) {
        const h16* b = Qh + (size_t)(q0 + n) * DK + c * 32;
        qf[c] = make_frag(*(const v8h*)(b + h8), *(const v8h*)(b + 16 + h8));
    }

    v8f o[16];
#pragma unroll
    for (int j = 0; j < 16; ++j) o[j] = {};
    float m_run[8], l_run[8];
#pragma unroll
    for (int v = 0; v < 8; ++v) { m_run[v] = -3.0e38f; l_run[v] = 0.f; }

    const float scale = 0.0625f;   // 1/sqrt(256)

    // ---- one full 32-key tile: 16 WMMA (S) + softmax + 16 WMMA (PV) ----
    auto compute_tile = [&](int kbase, int buf) {
        v8f s0 = {}, s1 = {};
#pragma unroll
        for (int c = 0; c < 8; ++c) {
            const h16* kb0 = &Kt[buf][(size_t)n * DK + c * 32];
            const h16* kb1 = &Kt[buf][(size_t)(16 + n) * DK + c * 32];
            v16h kA = make_frag(*(const v8h*)(kb0 + h8), *(const v8h*)(kb0 + 16 + h8));
            v16h kB = make_frag(*(const v8h*)(kb1 + h8), *(const v8h*)(kb1 + 16 + h8));
            s0 = __builtin_amdgcn_wmma_f32_16x16x32_f16(false, qf[c], false, kA, (short)0, s0, false, false);
            s1 = __builtin_amdgcn_wmma_f32_16x16x32_f16(false, qf[c], false, kB, (short)0, s1, false, false);
        }

        float alpha[8];
#pragma unroll
        for (int v = 0; v < 8; ++v) {
            const int qidx = old + q0 + h8 + v;
            float e0 = s0[v] * scale;
            float e1 = s1[v] * scale;
            if (kbase + n > qidx)      e0 = -__builtin_inff();
            if (kbase + 16 + n > qidx) e1 = -__builtin_inff();

            float t = fmaxf(e0, e1);
#pragma unroll
            for (int msk = 1; msk < 16; msk <<= 1)
                t = fmaxf(t, __shfl_xor(t, msk, 32));    // within 16-lane half-group

            const float mn = fmaxf(m_run[v], t);         // finite: no NaN on all-masked tiles
            const float a  = __expf(m_run[v] - mn);
            const float p0 = __expf(e0 - mn);
            const float p1 = __expf(e1 - mn);
            float rs = p0 + p1;
#pragma unroll
            for (int msk = 1; msk < 16; msk <<= 1)
                rs += __shfl_xor(rs, msk, 32);

            l_run[v] = l_run[v] * a + rs;
            m_run[v] = mn;
            alpha[v] = a;

            Pst[wave][h8 + v][n]      = (h16)p0;
            Pst[wave][h8 + v][16 + n] = (h16)p1;
        }

#pragma unroll
        for (int j = 0; j < 16; ++j)
#pragma unroll
            for (int v = 0; v < 8; ++v) o[j][v] *= alpha[v];

        // wave-local LDS RAW fence (same-wave DS ops are in-order)
        asm volatile("s_wait_dscnt 0" ::: "memory");
        v16h pf;
        {
            const h16* pb = &Pst[wave][n][0];
            pf = make_frag(*(const v8h*)(pb + h8), *(const v8h*)(pb + 16 + h8));
        }
        asm volatile("" ::: "memory");

#pragma unroll
        for (int j = 0; j < 16; ++j) {
            const h16* vbp = &Vts[buf][(size_t)(j * 16 + n) * 32];
            v16h vb = make_frag(*(const v8h*)(vbp + h8), *(const v8h*)(vbp + 16 + h8));
            o[j] = __builtin_amdgcn_wmma_f32_16x16x32_f16(false, pf, false, vb, (short)0, o[j], false, false);
        }
    };

#ifdef USE_TDM
    // ---------------- TDM-staged pipeline: one barrier per tile ----------------
    auto issue_tiles = [&](int buf, int kb) {
        const uint32_t remain = (uint32_t)(newN > kb ? newN - kb : 0);
        // K tile: 1-D contiguous 32*256 f16; OOB keys read back as zero
        tdm_load_2d(lds_offset(&Kt[buf][0]), Kh + (size_t)kb * DK,
                    remain * DK, 1u, 32u * DK, 0u, 0ull);
        // V tile: 2-D, 256 rows x 32 keys, row stride newN elements
        tdm_load_2d(lds_offset(&Vts[buf][0]), VtG + kb,
                    remain, (uint32_t)DV, 32u, (uint32_t)DV, (uint64_t)newN);
    };

    if (nIter > 0 && wave == 0) issue_tiles(0, k_lo);
    for (int it = 0; it < nIter; ++it) {
        const int kbase = k_lo + (it << 5);
        const int buf   = it & 1;
        if (wave == 0) __builtin_amdgcn_s_wait_tensorcnt(0);  // wave0's DMAs done
        __syncthreads();                                      // tile visible to all waves
        if (it + 1 < nIter && wave == 0) issue_tiles(buf ^ 1, kbase + 32);
        compute_tile(kbase, buf);
    }
#else
    // ---------------- fallback: register-staged double buffering ----------------
    v8h kreg[4], vreg[4];
    if (nIter > 0) {
#pragma unroll
        for (int i = 0; i < 4; ++i)
            kreg[i] = *(const v8h*)(Kh + (size_t)k_lo * DK + (size_t)(tid + i * 256) * 8);
#pragma unroll
        for (int i = 0; i < 4; ++i)
            vreg[i] = *(const v8h*)(VtG + (size_t)tid * newN + k_lo + i * 8);
#pragma unroll
        for (int i = 0; i < 4; ++i) *(v8h*)&Kt[0][(tid + i * 256) * 8] = kreg[i];
#pragma unroll
        for (int i = 0; i < 4; ++i) *(v8h*)&Vts[0][tid * 32 + i * 8] = vreg[i];
    }
    for (int it = 0; it < nIter; ++it) {
        const int kbase = k_lo + (it << 5);
        const int buf   = it & 1;
        __syncthreads();
        if (it + 1 < nIter) {
            const int kn = kbase + 32;
#pragma unroll
            for (int i = 0; i < 4; ++i)
                kreg[i] = *(const v8h*)(Kh + (size_t)kn * DK + (size_t)(tid + i * 256) * 8);
#pragma unroll
            for (int i = 0; i < 4; ++i)
                vreg[i] = *(const v8h*)(VtG + (size_t)tid * newN + kn + i * 8);
        }
        compute_tile(kbase, buf);
        __syncthreads();
        if (it + 1 < nIter) {
            const int nb = buf ^ 1;
#pragma unroll
            for (int i = 0; i < 4; ++i) *(v8h*)&Kt[nb][(tid + i * 256) * 8] = kreg[i];
#pragma unroll
            for (int i = 0; i < 4; ++i) *(v8h*)&Vts[nb][tid * 32 + i * 8] = vreg[i];
        }
    }
#endif

    // ---- epilogue: write unnormalized partials + (m, l) ----
#pragma unroll
    for (int v = 0; v < 8; ++v) {
        const int qrow = q0 + h8 + v;
        float* Op = Opart + ((size_t)qrow * NSEG + seg) * DV;
#pragma unroll
        for (int j = 0; j < 16; ++j) Op[j * 16 + n] = o[j][v];
        if (n == 0) {
            ML[((size_t)qrow * NSEG + seg) * 2 + 0] = m_run[v];
            ML[((size_t)qrow * NSEG + seg) * 2 + 1] = l_run[v];
        }
    }
}

// ---------------------------------------------------------------------------
// Combine pass: log-sum-exp merge of NSEG partials per query row.
// ---------------------------------------------------------------------------
__global__ void combine_partials(const float* __restrict__ Opart, const float* __restrict__ ML,
                                 float* __restrict__ out, int nQ) {
    int gid = blockIdx.x * blockDim.x + threadIdx.x;
    if (gid >= nQ * (DV / 4)) return;
    int q  = gid / (DV / 4);
    int c4 = (gid - q * (DV / 4)) * 4;

    float M = -3.0e38f;
#pragma unroll
    for (int s = 0; s < NSEG; ++s)
        M = fmaxf(M, ML[((size_t)q * NSEG + s) * 2]);

    float Ls = 0.f;
    float a0 = 0.f, a1 = 0.f, a2 = 0.f, a3 = 0.f;
#pragma unroll
    for (int s = 0; s < NSEG; ++s) {
        const float m_s = ML[((size_t)q * NSEG + s) * 2 + 0];
        const float l_s = ML[((size_t)q * NSEG + s) * 2 + 1];
        const float w   = __expf(m_s - M);
        Ls += l_s * w;
        float4 x = *(const float4*)(Opart + ((size_t)q * NSEG + s) * DV + c4);
        a0 += x.x * w; a1 += x.y * w; a2 += x.z * w; a3 += x.w * w;
    }
    const float inv = 1.0f / Ls;
    float4 r; r.x = a0 * inv; r.y = a1 * inv; r.z = a2 * inv; r.w = a3 * inv;
    *(float4*)(out + (size_t)q * DV + c4) = r;
}

// ---------------------------------------------------------------------------
// Host side
// ---------------------------------------------------------------------------
extern "C" void kernel_launch(void* const* d_in, const int* in_sizes, int n_in,
                              void* d_out, int out_size, void* d_ws, size_t ws_size,
                              hipStream_t stream) {
    const float* q   = (const float*)d_in[0];
    const float* k   = (const float*)d_in[1];
    const float* v   = (const float*)d_in[2];
    const float* Kc  = (const float*)d_in[3];
    const float* Vc  = (const float*)d_in[4];
    const int*   old = (const int*)d_in[5];

    const int Bq  = in_sizes[0] / DK;   // 2048
    const int mem = in_sizes[3] / DK;   // 32768 (worst-case new_size)

    // workspace layout (worst-case, host never reads old_size):
    char* ws = (char*)d_ws;
    size_t off = 0;
    h16* Qh  = (h16*)(ws + off); off += (size_t)Bq  * DK * 2;
    h16* Kh  = (h16*)(ws + off); off += (size_t)mem * DK * 2;
    h16* VtG = (h16*)(ws + off); off += (size_t)mem * DV * 2;
    float* Opart = (float*)(ws + off); off += (size_t)Bq * NSEG * DV * 4;
    float* ML    = (float*)(ws + off); off += (size_t)Bq * NSEG * 2 * 4;

    {
        int nElem = Bq * DK;
        build_q_f16<<<(nElem / 4 + 255) / 256, 256, 0, stream>>>(q, Qh, nElem);
    }
    {
        build_k_f16<<<(mem * DK / 4 + 255) / 256, 256, 0, stream>>>(Kc, k, old, Kh, Bq);
    }
    {
        dim3 g(mem / 16, DV / 16);
        dim3 b(16, 16);
        build_vt_f16<<<g, b, 0, stream>>>(Vc, v, old, VtG, Bq);
    }
    {
        dim3 g(Bq / 128, NSEG);   // 16 x 8 blocks, 8 waves each
        attn_flash_wmma<<<g, 256, 0, stream>>>(Qh, Kh, VtG, old, Opart, ML, Bq);
    }
    {
        int nThr = Bq * (DV / 4);
        combine_partials<<<(nThr + 255) / 256, 256, 0, stream>>>(Opart, ML, (float*)d_out, Bq);
    }
}
